// Head_52218212384824
// MI455X (gfx1250) — matrix operand order repaired and checked by reading, not compiled, not run on hardware
//
#include <hip/hip_runtime.h>

// ---------------- problem dims (match reference) ----------------
#define BN 8
#define TN 2048
#define CN 1024
#define HN 64

typedef __attribute__((ext_vector_type(16))) _Float16 v16h;
typedef __attribute__((ext_vector_type(8)))  _Float16 v8h;
typedef __attribute__((ext_vector_type(8)))  float    v8f;

static __device__ __forceinline__ v8f zero8() {
    v8f z;
#pragma unroll
    for (int i = 0; i < 8; ++i) z[i] = 0.0f;
    return z;
}

static __device__ __forceinline__ v16h cat8(v8h lo, v8h hi) {
    v16h r;
#pragma unroll
    for (int i = 0; i < 8; ++i) { r[i] = lo[i]; r[i + 8] = hi[i]; }
    return r;
}

// D = A*B + C, 16x16 tile, K=32, f16 inputs, f32 accumulate.
static __device__ __forceinline__ v8f wmma16(v16h a, v16h b, v8f c) {
    return __builtin_amdgcn_wmma_f32_16x16x32_f16(
        false, a, false, b, (short)0, c, false, false);
}

// A-operand fragment: 16x32 (MxK) f16 tile, row-major source, leading dim ld.
// lane<16 -> row=lane, K={0..7},{16..23}; lane>=16 -> row=lane-16, K={8..15},{24..31}.
static __device__ __forceinline__ v16h load_a_f16(const _Float16* src, int ld, int lane) {
    int r = lane & 15, h = lane >> 4;
    const _Float16* p = src + r * ld + 8 * h;
    v8h lo = *(const v8h*)(p);
    v8h hi = *(const v8h*)(p + 16);
    return cat8(lo, hi);
}

// B-operand fragment: 32x16 (KxN) with B[k][n] = src[n*ld + k]
// (each of the 16 N-columns stored contiguously along K).
static __device__ __forceinline__ v16h load_b_cols(const _Float16* src, int ld, int lane) {
    int n = lane & 15, kb = (lane >> 4) * 8;
    const _Float16* p = src + n * ld + kb;
    v8h lo = *(const v8h*)(p);
    v8h hi = *(const v8h*)(p + 16);
    return cat8(lo, hi);
}

// A-operand fragment from an f32 row-major source (converted to f16 on load).
static __device__ __forceinline__ v16h load_a_f32(const float* src, int ld, int lane) {
    int r = lane & 15, h = lane >> 4;
    const float* p = src + (size_t)r * ld + 8 * h;
    v16h out;
#pragma unroll
    for (int i = 0; i < 8; ++i) out[i] = (_Float16)p[i];
#pragma unroll
    for (int i = 0; i < 8; ++i) out[i + 8] = (_Float16)p[16 + i];
    return out;
}

// gfx1250 async DMA: memory -> LDS, 16B per lane, tracked by ASYNCcnt.
// lds_addr: LDS byte address (low 32 bits of flat shared addr).
static __device__ __forceinline__ void async_b128(unsigned lds_addr, unsigned voff,
                                                  unsigned long long saddr) {
    asm volatile("global_load_async_to_lds_b128 %0, %1, %2"
                 :: "v"(lds_addr), "v"(voff), "s"(saddr) : "memory");
}
#define WAIT_ASYNC_LE(n) asm volatile("s_wait_asynccnt " n ::: "memory")

// ============================================================================
// Kernel 0: one-shot W convert + transpose: Wt[m][col(64)][k(1024)] f16.
// Makes the projection kernel's W staging a raw async byte copy.
// ============================================================================
__global__ __launch_bounds__(256) void wcvt_kernel(
    const float* __restrict__ Wk, const float* __restrict__ Wq,
    const float* __restrict__ Wv, _Float16* __restrict__ wt)
{
    int idx = blockIdx.x * 256 + threadIdx.x;      // 0 .. 3*64*1024-1
    int m   = idx >> 16;
    int col = (idx >> 10) & 63;
    int kk  = idx & 1023;
    const float* W = (m == 0) ? Wk : (m == 1) ? Wq : Wv;
    wt[idx] = (_Float16)W[kk * HN + col];
}

// ============================================================================
// Kernel 1: q/k/v projections.  x:[B*T, C] f32 @ Wt -> f16 ws.
// 8 waves/block, wave owns a 16-row x tile. Pre-transposed W chunks are
// double-buffered into LDS with async DMA (3 x 512B per wave per chunk).
// V is written TRANSPOSED (Vt[b][hd][t]) so attention staging is a raw copy.
// ============================================================================
__global__ __launch_bounds__(256) void proj_qkv_kernel(
    const float* __restrict__ x,
    const _Float16* __restrict__ wt,
    _Float16* __restrict__ kout,
    _Float16* __restrict__ qout,
    _Float16* __restrict__ vtout)
{
    __shared__ _Float16 ldsW[2][3 * 64 * 32];   // [buf][m][col(64)][kk(32)]

    const int tid  = threadIdx.x;
    const int lane = tid & 31;
    const int wave = tid >> 5;
    const int n16  = lane & 15;
    const int half = lane >> 4;

    const size_t row0 = ((size_t)blockIdx.x * 8 + wave) * 16;   // 16-row x tile

    const unsigned ldsWb[2] = { (unsigned)(uintptr_t)&ldsW[0][0],
                                (unsigned)(uintptr_t)&ldsW[1][0] };
    const unsigned long long wt_g = (unsigned long long)(uintptr_t)wt;

    // stage chunk c (Wt[m][col][32c..32c+32], 3 x 4KB) into buffer (c&1)
    auto stage = [&](int c) {
#pragma unroll
        for (int m = 0; m < 3; ++m) {
            unsigned voff = (unsigned)(m * 131072 + (tid >> 2) * 2048 +
                                       c * 64 + (tid & 3) * 16);
            async_b128(ldsWb[c & 1] + (unsigned)(m * 4096 + tid * 16), voff, wt_g);
        }
    };

    v8f acc[3][4];
#pragma unroll
    for (int m = 0; m < 3; ++m)
#pragma unroll
        for (int n = 0; n < 4; ++n) acc[m][n] = zero8();

    stage(0);
    for (int c = 0; c < CN / 32; ++c) {
        if (c + 1 < CN / 32) { stage(c + 1); WAIT_ASYNC_LE("0x3"); }
        else                 { WAIT_ASYNC_LE("0x0"); }
        __syncthreads();

        const _Float16* Wc = &ldsW[c & 1][0];
        v16h afrag = load_a_f32(x + row0 * CN + c * 32, CN, lane);
#pragma unroll
        for (int m = 0; m < 3; ++m)
#pragma unroll
            for (int n = 0; n < 4; ++n) {
                v16h bfrag = load_b_cols(Wc + m * 2048 + (n * 16) * 32, 32, lane);
                acc[m][n] = wmma16(afrag, bfrag, acc[m][n]);
            }
        __syncthreads();   // readers done before next stage overwrites other buf
    }

    const size_t bb = row0 / TN, t0 = row0 % TN;
#pragma unroll
    for (int n = 0; n < 4; ++n)
#pragma unroll
        for (int r = 0; r < 8; ++r) {
            int row = r + 8 * half;
            int col = n * 16 + n16;
            kout[(row0 + row) * HN + col] = (_Float16)acc[0][n][r];
            qout[(row0 + row) * HN + col] = (_Float16)acc[1][n][r];
            vtout[(bb * HN + col) * TN + t0 + row] = (_Float16)acc[2][n][r];
        }
}

// ============================================================================
// Kernel 2: causal flash attention. 8 waves/block = 128 query rows, one batch.
// K (row-major) and Vt (pre-transposed) tiles double-buffered via async DMA.
// ============================================================================
__global__ __launch_bounds__(256) void flash_attn_kernel(
    const _Float16* __restrict__ qbf,
    const _Float16* __restrict__ kbf,
    const _Float16* __restrict__ vtbf,
    float* __restrict__ out)
{
    __shared__ _Float16 ldsK[2][32 * HN];    // key tile [32 x 64]
    __shared__ _Float16 ldsVt[2][HN * 32];   // value tile transposed [64 x 32]
    __shared__ _Float16 ldsP[8][16 * 32];    // per-wave P scratch (D->A relayout)

    const int tid  = threadIdx.x;
    const int lane = tid & 31;
    const int wave = tid >> 5;
    const int n16  = lane & 15;
    const int half = lane >> 4;

    const int b    = blockIdx.y;
    const int q0wg = blockIdx.x * 128;
    const int q0   = q0wg + wave * 16;

    const _Float16* qsrc = qbf + ((size_t)b * TN + q0) * HN;
    const v16h qa0 = load_a_f16(qsrc + 0,  HN, lane);
    const v16h qa1 = load_a_f16(qsrc + 32, HN, lane);

    v8f o0 = zero8(), o1 = zero8(), o2 = zero8(), o3 = zero8();
    float m_r[8], l_r[8];
#pragma unroll
    for (int r = 0; r < 8; ++r) { m_r[r] = -__builtin_inff(); l_r[r] = 0.0f; }

    const float sc = 0.03125f;               // C^-0.5 = 1/32
    const int nblocks = q0wg / 32 + 4;

    const unsigned ldsKb[2] = { (unsigned)(uintptr_t)&ldsK[0][0],
                                (unsigned)(uintptr_t)&ldsK[1][0] };
    const unsigned ldsVb[2] = { (unsigned)(uintptr_t)&ldsVt[0][0],
                                (unsigned)(uintptr_t)&ldsVt[1][0] };
    const unsigned long long k_g = (unsigned long long)(uintptr_t)(kbf + (size_t)b * TN * HN);
    const unsigned long long v_g = (unsigned long long)(uintptr_t)(vtbf + (size_t)b * HN * TN);

    auto stage = [&](int j) {
        const int kb = j * 32;
        // K tile: contiguous 4KB, 512B per wave
        async_b128(ldsKb[j & 1] + (unsigned)(tid * 16),
                   (unsigned)(kb * HN * 2 + tid * 16), k_g);
        // Vt tile: 64 rows (head dim) x 32 keys; row = tid>>2, 16B sub-chunk
        async_b128(ldsVb[j & 1] + (unsigned)(tid * 16),
                   (unsigned)(((tid >> 2) * TN + kb + (tid & 3) * 8) * 2), v_g);
    };

    stage(0);
    for (int j = 0; j < nblocks; ++j) {
        const int kb = j * 32;
        if (j + 1 < nblocks) { stage(j + 1); WAIT_ASYNC_LE("0x2"); }
        else                 { WAIT_ASYNC_LE("0x0"); }
        __syncthreads();

        if (kb <= q0 + 15) {                 // wave-uniform causal skip
            const _Float16* Kc = &ldsK[j & 1][0];
            const _Float16* Vc = &ldsVt[j & 1][0];

            // ---- S = q @ k^T ----
            v8f s0 = zero8(), s1 = zero8();
            s0 = wmma16(qa0, load_b_cols(Kc + 0 * HN + 0,  HN, lane), s0);
            s0 = wmma16(qa1, load_b_cols(Kc + 0 * HN + 32, HN, lane), s0);
            s1 = wmma16(qa0, load_b_cols(Kc + 16 * HN + 0,  HN, lane), s1);
            s1 = wmma16(qa1, load_b_cols(Kc + 16 * HN + 32, HN, lane), s1);

            // ---- online softmax ----
            float pr0[8], pr1[8];
#pragma unroll
            for (int r = 0; r < 8; ++r) {
                const int trow = q0 + r + 8 * half;
                float a = s0[r] * sc;
                float c = s1[r] * sc;
                if (kb + n16 > trow)      a = -__builtin_inff();
                if (kb + 16 + n16 > trow) c = -__builtin_inff();
                float mx = fmaxf(a, c);
                mx = fmaxf(mx, __shfl_xor(mx, 1, 32));
                mx = fmaxf(mx, __shfl_xor(mx, 2, 32));
                mx = fmaxf(mx, __shfl_xor(mx, 4, 32));
                mx = fmaxf(mx, __shfl_xor(mx, 8, 32));
                const float mnew = fmaxf(m_r[r], mx);
                const float p0 = __expf(a - mnew);
                const float p1 = __expf(c - mnew);
                float rs = p0 + p1;
                rs += __shfl_xor(rs, 1, 32);
                rs += __shfl_xor(rs, 2, 32);
                rs += __shfl_xor(rs, 4, 32);
                rs += __shfl_xor(rs, 8, 32);
                const float corr = __expf(m_r[r] - mnew);
                l_r[r] = l_r[r] * corr + rs;
                m_r[r] = mnew;
                o0[r] *= corr; o1[r] *= corr; o2[r] *= corr; o3[r] *= corr;
                pr0[r] = p0; pr1[r] = p1;
            }

            // ---- relayout P (f32 D-layout) -> f16 A-layout via per-wave LDS ----
            _Float16* pp = &ldsP[wave][0];
#pragma unroll
            for (int r = 0; r < 8; ++r) {
                int row = r + 8 * half;
                pp[row * 32 + n16]      = (_Float16)pr0[r];
                pp[row * 32 + 16 + n16] = (_Float16)pr1[r];
            }
            v16h pa = load_a_f16(pp, 32, lane);

            // ---- O += P @ V ----
            o0 = wmma16(pa, load_b_cols(Vc + 0  * 32, 32, lane), o0);
            o1 = wmma16(pa, load_b_cols(Vc + 16 * 32, 32, lane), o1);
            o2 = wmma16(pa, load_b_cols(Vc + 32 * 32, 32, lane), o2);
            o3 = wmma16(pa, load_b_cols(Vc + 48 * 32, 32, lane), o3);
        }
        __syncthreads();                     // readers done before next stage
    }

    // ---- epilogue: normalize, store fp32 [B, T, 64] ----
#pragma unroll
    for (int r = 0; r < 8; ++r) {
        const float inv = 1.0f / l_r[r];
        const int row = q0 + r + 8 * half;
        float* op = out + ((size_t)b * TN + row) * HN + n16;
        op[0]  = o0[r] * inv;
        op[16] = o1[r] * inv;
        op[32] = o2[r] * inv;
        op[48] = o3[r] * inv;
    }
}

// ============================================================================
extern "C" void kernel_launch(void* const* d_in, const int* in_sizes, int n_in,
                              void* d_out, int out_size, void* d_ws, size_t ws_size,
                              hipStream_t stream) {
    (void)in_sizes; (void)n_in; (void)out_size; (void)ws_size;
    const float* x  = (const float*)d_in[0];
    const float* Wk = (const float*)d_in[1];
    const float* Wq = (const float*)d_in[2];
    const float* Wv = (const float*)d_in[3];
    float* out = (float*)d_out;

    const size_t per = (size_t)BN * TN * HN;       // 1M f16 elems each
    _Float16* kbf  = (_Float16*)d_ws;
    _Float16* qbf  = kbf + per;
    _Float16* vtbf = qbf + per;                    // transposed V: [B][H][T]
    _Float16* wt   = vtbf + per;                   // transposed W: [3][64][1024]

    // 0) W -> f16, pre-transposed (3*64*1024 elems).
    wcvt_kernel<<<dim3(768), 256, 0, stream>>>(Wk, Wq, Wv, wt);

    // 1) q/k/v = x @ W (f32 -> f16 ws; V stored transposed).
    proj_qkv_kernel<<<dim3(128), 256, 0, stream>>>(x, wt, kbf, qbf, vtbf);

    // 2) causal flash attention over f16 q/k/vt, fp32 output.
    flash_attn_kernel<<<dim3(TN / 128, BN), 256, 0, stream>>>(qbf, kbf, vtbf, out);
}
